// SubtractorNBit_43860206027335
// MI455X (gfx1250) — matrix operand order repaired and checked by reading, not compiled, not run on hardware
//
#include <hip/hip_runtime.h>

// 16-bit ripple-borrow subtractor over {0,1} f32 spikes.
//   diff = (A - B) mod 2^16 (bitwise), borrow = (A < B)
// One v_wmma_f32_16x16x32_bf16 per 16 batch rows:
//   K=0..15  = A bits, weight +2^k
//   K=16..31 = B bits, weight -2^k
//   -> every D column = uA(m) - uB(m), exact integer in f32.

typedef __attribute__((ext_vector_type(16))) __bf16 v16bf;
typedef __attribute__((ext_vector_type(8)))  float  v8f;

#define BITS            16
#define BATCH           2097152
#define ROWS_PER_TILE   16
#define TILES_TOTAL     (BATCH / ROWS_PER_TILE)   // 131072
#define TILES_PER_WAVE  8
#define WAVES_PER_BLOCK 8
#define BLOCK_THREADS   256
#define GRID_BLOCKS     (TILES_TOTAL / (TILES_PER_WAVE * WAVES_PER_BLOCK)) // 2048

union BF16Vec { v16bf v; unsigned int u[8]; };

// inputs are exactly 0.0f / 1.0f -> bf16 bits == top 16 bits of the f32 word.
// v_perm_b32: result = [hi.b3, hi.b2, lo.b3, lo.b2]
__device__ __forceinline__ unsigned int pack_hi16(float lo, float hi) {
    return __builtin_amdgcn_perm(__float_as_uint(hi), __float_as_uint(lo), 0x07060302u);
}

__global__ __launch_bounds__(BLOCK_THREADS)
void sub16_wmma_kernel(const float* __restrict__ A, const float* __restrict__ B,
                       float* __restrict__ outDiff, float* __restrict__ outBorrow) {
    const int lane  = threadIdx.x & 31;
    const int gwave = blockIdx.x * WAVES_PER_BLOCK + (threadIdx.x >> 5);
    const int half  = lane >> 4;   // K-half this lane supplies / D-row block it holds
    const int s     = lane & 15;   // row loaded by this lane; also bit column stored

    // B operand (weights), lane-resident constants:
    // table order: elems 0-7 hold K=8h+e -> +2^k ; elems 8-15 hold K=16+8h+(e-8) -> -2^(k-16)
    BF16Vec wb;
#pragma unroll
    for (int j = 0; j < 8; ++j) {
        unsigned int lo = ((unsigned int)(127 + 8 * half + 2 * (j & 3)) << 7) | ((unsigned int)(j >> 2) << 15);
        unsigned int hi = ((unsigned int)(128 + 8 * half + 2 * (j & 3)) << 7) | ((unsigned int)(j >> 2) << 15);
        wb.u[j] = lo | (hi << 16);
    }

    const size_t rowbase0 = (size_t)gwave * (TILES_PER_WAVE * ROWS_PER_TILE);
    const float* pA = A + (rowbase0 + s) * BITS + 8 * half;  // this lane's 8 A bits
    const float* pB = B + (rowbase0 + s) * BITS + 8 * half;  // this lane's 8 B bits
    float*       pD = outDiff + (rowbase0 + (size_t)(8 * half)) * BITS + s; // bit col s
    float*       pR = outBorrow + rowbase0 + 8 * half + ((lane >> 3) & 1) * 4;
    const int    q  = (lane >> 3) & 1;     // which di quad feeds the borrow store

    for (int i = 0; i < TILES_PER_WAVE; ++i) {
        float4 a0 = ((const float4*)pA)[0], a1 = ((const float4*)pA)[1];
        float4 b0 = ((const float4*)pB)[0], b1 = ((const float4*)pB)[1];

        // A operand: VGPR0-3 = 8 A bits (K=0..7 of this lane's K-half),
        //            VGPR4-7 = 8 B bits (K=16..23 of this lane's K-half)
        BF16Vec am;
        am.u[0] = pack_hi16(a0.x, a0.y);
        am.u[1] = pack_hi16(a0.z, a0.w);
        am.u[2] = pack_hi16(a1.x, a1.y);
        am.u[3] = pack_hi16(a1.z, a1.w);
        am.u[4] = pack_hi16(b0.x, b0.y);
        am.u[5] = pack_hi16(b0.z, b0.w);
        am.u[6] = pack_hi16(b1.x, b1.y);
        am.u[7] = pack_hi16(b1.z, b1.w);

        v8f c = {};
        v8f d = __builtin_amdgcn_wmma_f32_16x16x32_bf16(
            /*neg_a=*/false, am.v, /*neg_b=*/false, wb.v,
            /*c_mod=*/(short)0, c, /*reuse_a=*/false, /*reuse_b=*/false);

        // Slot j holds row (8*half + j); identical across the 16 lanes of a half.
        int di[8];
#pragma unroll
        for (int j = 0; j < 8; ++j)
            di[j] = (int)d[j];                       // ua - ub, exact integer

        // Iteration j: the 16 lanes of this half write the 16 diff bits of
        // row 8*half + j; lane stores bit s (v_bfe + v_cvt + b32 store).
#pragma unroll
        for (int j = 0; j < 8; ++j) {
            unsigned int bit = ((unsigned int)di[j] >> s) & 1u;
            pD[j * BITS] = (float)bit;
        }

        // Borrow: lanes {0,8,16,24} each write 4 consecutive borrows as b128.
        int e0 = q ? di[4] : di[0];
        int e1 = q ? di[5] : di[1];
        int e2 = q ? di[6] : di[2];
        int e3 = q ? di[7] : di[3];
        if ((lane & 7) == 0) {
            *(float4*)pR = make_float4(e0 < 0 ? 1.0f : 0.0f, e1 < 0 ? 1.0f : 0.0f,
                                       e2 < 0 ? 1.0f : 0.0f, e3 < 0 ? 1.0f : 0.0f);
        }

        pA += ROWS_PER_TILE * BITS;
        pB += ROWS_PER_TILE * BITS;
        pD += ROWS_PER_TILE * BITS;
        pR += ROWS_PER_TILE;
    }
}

extern "C" void kernel_launch(void* const* d_in, const int* in_sizes, int n_in,
                              void* d_out, int out_size, void* d_ws, size_t ws_size,
                              hipStream_t stream) {
    (void)in_sizes; (void)n_in; (void)out_size; (void)d_ws; (void)ws_size;
    const float* A = (const float*)d_in[0];
    const float* B = (const float*)d_in[1];
    float* outDiff   = (float*)d_out;                          // [BATCH, 16]
    float* outBorrow = (float*)d_out + (size_t)BATCH * BITS;   // [BATCH, 1]
    sub16_wmma_kernel<<<GRID_BLOCKS, BLOCK_THREADS, 0, stream>>>(A, B, outDiff, outBorrow);
}
